// Wavelet1D_45183055954444
// MI455X (gfx1250) — matrix operand order repaired and checked by reading, not compiled, not run on hardware
//
#include <hip/hip_runtime.h>
#include <hip/hip_bf16.h>
#include <math.h>
#include <stdint.h>

typedef __attribute__((ext_vector_type(16))) _Float16 v16h;
typedef __attribute__((ext_vector_type(8)))  float    v8f;
typedef __attribute__((ext_vector_type(4)))  unsigned int u32x4;
typedef __attribute__((ext_vector_type(8)))  unsigned int u32x8;

// ---------------------------------------------------------------------------
// Problem constants (from reference)
// ---------------------------------------------------------------------------
#define BB 8
#define CC 64
#define HH 128
#define WW 128
#define HWHW 16384         // H*W
#define OO 64
#define KK 6
#define GG 16
#define TILE 128           // pixels per block
#define THREADS 256

// Raw (un-normalized) wavelet low-pass filters; normalized on device.
__device__ const float F_DB4[8] = {
  -0.0105974017850021f, 0.0328830116668852f, 0.0308413818355607f, -0.1870348117188811f,
  -0.0279837694169839f, 0.6308807679295904f, 0.7148465705529155f,  0.2303778133088964f };
__device__ const float F_DB6[12] = {
   0.00107730108499558f, -0.00477725751101065f, -0.0005538422009938f, 0.03158203931748603f,
   0.02752286553030533f, -0.0975016055873225f, -0.12976686756709563f, 0.22626469396544f,
   0.3152503517092432f,  -0.7511339080210959f,  0.4946238903984534f,  0.1115407433501095f };
__device__ const float F_SYM6[12] = {
  -0.007800708325034148f, 0.001767711864242804f, 0.04472490177066578f, -0.02106029251230056f,
  -0.0726375227866f,      0.3379294217282401f,   0.787641141030194f,    0.4910559419267466f,
  -0.048311742585632f,   -0.1179901111484105f,   0.00349071208421747f,  0.01540410932702737f };
__device__ const float F_COIF5[30] = {
  -3.459977283621256e-05f, -7.098330313814114e-05f, 0.0004662169601128863f, 0.001117518770890601f,
  -0.002574517688750223f, -0.00900797613666158f, 0.015880544863615904f, 0.03455502757306163f,
  -0.08230192710688598f, -0.07179982161931202f, 0.42848347637761874f, 0.7937772226256206f,
   0.4051769024096169f, -0.06112339000267287f, -0.06577191128185562f, 0.023452696141836267f,
   0.007782596427325418f, -0.003793512864491014f, -0.0002606761356811993f, 0.000107502882505652f,
   1.10319778524429e-05f, -5.520763127949e-06f, -1.0682196848076e-06f, 5.236425333584e-07f,
   1.125098976034e-07f, -5.417490769329e-08f, -8.8631e-09f, 4.2921e-09f, 6.7e-10f, -3.2e-10f };

// linear resize (exact replica of reference _resize_linear) src[Lin] -> dst[64]
__device__ __forceinline__ void resize_to64(const float* src, int Lin, float* dst) {
  const float scale = (float)Lin / 64.0f;
  for (int i = 0; i < 64; ++i) {
    float s = (i + 0.5f) * scale - 0.5f;
    if (s < 0.0f) s = 0.0f;
    int i0 = (int)floorf(s);
    if (i0 > Lin - 1) i0 = Lin - 1;
    int i1 = i0 + 1; if (i1 > Lin - 1) i1 = Lin - 1;
    float w = s - (float)i0;
    dst[i] = src[i0] * (1.0f - w) + src[i1] * w;
  }
}

// neighbor sum with reflect padding on length-32 signal
__device__ __forceinline__ void nsum32(const float* y, float* out) {
  for (int i = 0; i < 32; ++i) {
    int l = (i == 0)  ? 1  : i - 1;
    int r = (i == 31) ? 30 : i + 1;
    out[i] = 0.5f * (y[l] + y[r]);
  }
}

// ---------------------------------------------------------------------------
// Setup: build Q_k = P_A * M_k^A + P_D * M_k^D  (6 matrices, 64x64, f16)
// Q layout: Q[(k*64 + o)*64 + j]
// ---------------------------------------------------------------------------
__global__ void build_q_kernel(const float* __restrict__ proj_w,
                               const float* __restrict__ s53A, const float* __restrict__ s53D,
                               const float* __restrict__ s97A, const float* __restrict__ s97D,
                               _Float16* __restrict__ Q) {
  const int k = blockIdx.x;     // transform id 0..5
  const int j = threadIdx.x;    // basis column 0..63
  if (j >= 64) return;

  float A[64], D[64];

  if (k < 4) {
    const float* lo; int L;
    if      (k == 0) { lo = F_DB4;  L = 8;  }
    else if (k == 1) { lo = F_DB6;  L = 12; }
    else if (k == 2) { lo = F_SYM6; L = 12; }
    else             { lo = F_COIF5; L = 30; }
    float nrm = 0.0f;
    for (int t = 0; t < L; ++t) nrm += lo[t] * lo[t];
    nrm = sqrtf(nrm) + 1e-12f;
    const int pad = (L - 1) / 2;     // L even here
    const int pl  = pad - 1;
    float a62[62], d62[62];
    for (int i = 0; i < 62; ++i) {
      float sa = 0.0f, sd = 0.0f;
      for (int t = 0; t < L; ++t) {
        int m = i + t - pl;
        if (m < 0)  m = -m;            // reflect (no edge repeat)
        if (m > 63) m = 126 - m;
        if (m == j) {                  // x = basis vector e_j
          float lon = lo[t] / nrm;
          float hin = ((t & 1) ? -1.0f : 1.0f) * (lo[L - 1 - t] / nrm);
          sa += lon; sd += hin;
        }
      }
      a62[i] = sa; d62[i] = sd;
    }
    resize_to64(a62, 62, A);
    resize_to64(d62, 62, D);
  } else {
    float ev[32], od[32], t[32];
    for (int i = 0; i < 32; ++i) {
      ev[i] = (2 * i     == j) ? 1.0f : 0.0f;
      od[i] = (2 * i + 1 == j) ? 1.0f : 0.0f;
    }
    float sA, sD;
    if (k == 4) {
      nsum32(ev, t); for (int i = 0; i < 32; ++i) od[i] += -0.5f * t[i];
      nsum32(od, t); for (int i = 0; i < 32; ++i) ev[i] +=  0.25f * t[i];
      sA = *s53A; sD = *s53D;
    } else {
      const float al = -1.586134342f, be = -0.05298011854f, ga = 0.8829110762f, de = 0.4435068522f;
      nsum32(ev, t); for (int i = 0; i < 32; ++i) od[i] += al * t[i];
      nsum32(od, t); for (int i = 0; i < 32; ++i) ev[i] += be * t[i];
      nsum32(ev, t); for (int i = 0; i < 32; ++i) od[i] += ga * t[i];
      nsum32(od, t); for (int i = 0; i < 32; ++i) ev[i] += de * t[i];
      sA = *s97A; sD = *s97D;
    }
    float a32[32], d32[32];
    for (int i = 0; i < 32; ++i) { a32[i] = sA * ev[i]; d32[i] = sD * od[i]; }
    resize_to64(a32, 32, A);
    resize_to64(d32, 32, D);
  }

  // Q_k[:, j] = P_A * A + P_D * D  (proj_w is (64, 128) row-major)
  for (int o = 0; o < 64; ++o) {
    float q = 0.0f;
    const float* pw = proj_w + o * 128;
    for (int i = 0; i < 64; ++i) q += pw[i] * A[i] + pw[64 + i] * D[i];
    Q[((k * 64 + o) << 6) + j] = (_Float16)q;
  }
}

// ---------------------------------------------------------------------------
// Main fused kernel: TDM staging + gate MLP + 6x(64x64) WMMA GEMM + mix
// ---------------------------------------------------------------------------
// dynamic LDS layout (bytes)
#define OFF_XC   0                    // f32 x tile [c=64][p=128] (TDM dest), reused as OutS [64][130]
#define OFF_QH   33280                // f16 Qstack [384][72]  (TDM dest with padding)
#define OFF_WM   88576                // f32 wm    [128][8]
#define OFF_W1   92672                // f32 gate_w1 [16][64]
#define OFF_W2   96768                // f32 gate_w2 [6][16]
#define OFF_B1   97152
#define OFF_B2   97216
#define OFF_PB   97248
#define SMEM_BYTES 97504

union AFrag { v16h h; unsigned int u[8]; };

// Issue a 2D TDM tensor load (VADDR2/VADDR3 = NULL) described by D# groups 0/1.
__device__ __forceinline__ void tdm_load_2d(u32x4 g0, u32x8 g1) {
  asm volatile("tensor_load_to_lds %0, %1" :: "s"(g0), "s"(g1) : "memory");
}

__global__ __launch_bounds__(THREADS) void wavelet_main_kernel(
    const float* __restrict__ x,
    const float* __restrict__ gw1, const float* __restrict__ gb1,
    const float* __restrict__ gw2, const float* __restrict__ gb2,
    const float* __restrict__ pb,
    const _Float16* __restrict__ Qws,
    float* __restrict__ out) {
  extern __shared__ char smem[];
  float*    Xcp  = (float*)(smem + OFF_XC);     // [c][128]
  float*    OutS = (float*)(smem + OFF_XC);     // [o][130] (aliases Xcp)
  _Float16* Qh   = (_Float16*)(smem + OFF_QH);  // [k*64+o][72]
  float*    wmS  = (float*)(smem + OFF_WM);     // [p][8]
  float*    W1s  = (float*)(smem + OFF_W1);
  float*    W2s  = (float*)(smem + OFF_W2);
  float*    b1s  = (float*)(smem + OFF_B1);
  float*    b2s  = (float*)(smem + OFF_B2);
  float*    pbs  = (float*)(smem + OFF_PB);

  const int tid   = threadIdx.x;
  const int n0    = blockIdx.x * TILE;
  const int b     = n0 >> 14;          // / (H*W)
  const int plane = n0 & (HWHW - 1);

  // ---- TDM staging: x tile (64 rows x 128 f32) and Qstack (384 rows x 64 f16,
  //      hardware-padded to 72-half LDS rows). Issued by wave 0 only (TDM
  //      ignores EXEC; branch keeps other waves from re-issuing). ----
  if (tid < 32) {
    {
      uint64_t ga = (uint64_t)(uintptr_t)(x + (size_t)b * CC * HWHW + plane);
      uint32_t lds = (uint32_t)(uintptr_t)Xcp;
      u32x4 g0 = { 1u,                                   // count=1, no gather
                   lds,                                  // lds_addr
                   (uint32_t)ga,                         // global_addr[31:0]
                   (uint32_t)((ga >> 32) & 0x01FFFFFFu) | 0x80000000u }; // addr[56:32] | type=2
      u32x8 g1 = { (2u << 16),                           // data_size=4B
                   (uint32_t)(HWHW & 0xFFFF) << 16,      // tensor_dim0 lo16 (16384)
                   (uint32_t)(HWHW >> 16) | (64u << 16), // tensor_dim0 hi16 | tensor_dim1 lo16
                   (128u << 16),                         // tensor_dim1 hi16 | tile_dim0=128
                   64u,                                  // tile_dim1=64 | tile_dim2=0
                   (uint32_t)HWHW,                       // tensor_dim0_stride lo32 = 16384
                   0u, 0u };
      tdm_load_2d(g0, g1);
    }
    {
      uint64_t ga = (uint64_t)(uintptr_t)Qws;
      uint32_t lds = (uint32_t)(uintptr_t)Qh;
      u32x4 g0 = { 1u, lds, (uint32_t)ga,
                   (uint32_t)((ga >> 32) & 0x01FFFFFFu) | 0x80000000u };
      // data_size=2B, pad_enable, pad_interval=4 (32 DWORDs=128B rows),
      // pad_amount=3 (4 DWORDs=16B -> 72-half pitch)
      u32x8 g1 = { (1u << 16) | (1u << 20) | (4u << 22) | (3u << 25),
                   (64u << 16),              // tensor_dim0 = 64
                   (384u << 16),             // tensor_dim1 = 384
                   (64u << 16),              // tile_dim0 = 64
                   384u,                     // tile_dim1 = 384
                   64u,                      // tensor_dim0_stride = 64
                   0u, 0u };
      tdm_load_2d(g0, g1);
    }
    __builtin_amdgcn_s_wait_tensorcnt(0);
  }
  // ---- stage gate weights / biases (tiny) ----
  for (int i = tid; i < GG * CC; i += THREADS) W1s[i] = gw1[i];
  if (tid < KK * GG) W2s[tid] = gw2[tid];
  if (tid < GG) b1s[tid] = gb1[tid];
  if (tid < KK) b2s[tid] = gb2[tid];
  if (tid < OO) pbs[tid] = pb[tid];
  __syncthreads();

  // ---- gate MLP + softmax (one pixel per thread, threads 0..127) ----
  if (tid < TILE) {
    float h[GG];
#pragma unroll
    for (int g = 0; g < GG; ++g) h[g] = b1s[g];
    for (int c = 0; c < CC; ++c) {
      float xv = Xcp[c * 128 + tid];
#pragma unroll
      for (int g = 0; g < GG; ++g) h[g] += W1s[g * CC + c] * xv;
    }
#pragma unroll
    for (int g = 0; g < GG; ++g) h[g] = fmaxf(h[g], 0.0f);
    float lg[KK], mx = -3.0e38f;
#pragma unroll
    for (int k = 0; k < KK; ++k) {
      float s = b2s[k];
#pragma unroll
      for (int g = 0; g < GG; ++g) s += W2s[k * GG + g] * h[g];
      lg[k] = s; mx = fmaxf(mx, s);
    }
    float e[KK], ssum = 0.0f;
#pragma unroll
    for (int k = 0; k < KK; ++k) { e[k] = expf(lg[k] - mx); ssum += e[k]; }
    float inv = 1.0f / ssum, s2 = 0.0f;
#pragma unroll
    for (int k = 0; k < KK; ++k) { e[k] *= inv; s2 += e[k]; }
    float inv2 = 1.0f / (s2 + 1e-12f);
#pragma unroll
    for (int k = 0; k < KK; ++k) wmS[tid * 8 + k] = e[k] * inv2;
  }
  __syncthreads();

  // ---- WMMA: each wave owns a 16-pixel stripe ----
  const int wv    = tid >> 5;
  const int lane  = tid & 31;
  const int lrow  = lane & 15;
  const int khalf = lane >> 4;
  const int row0  = wv * 16;

  // A fragments (16x64 f16 = 2 chunks of 16x32), built from f32 channel-major
  // tile with on-the-fly f16 convert. ISA A layout:
  //   K = 16*(r>=4) + 8*khalf + 2*(r&3) (+lo/hi half within the b32 pair)
  AFrag afr[2];
  {
    const float* xc = Xcp + (row0 + lrow);   // fixed pixel column
#pragma unroll
    for (int kc = 0; kc < 2; ++kc) {
#pragma unroll
      for (int r = 0; r < 8; ++r) {
        int K = kc * 32 + ((r >> 2) << 4) + (khalf << 3) + ((r & 3) << 1);
        afr[kc].h[2 * r]     = (_Float16)xc[K * 128];
        afr[kc].h[2 * r + 1] = (_Float16)xc[(K + 1) * 128];
      }
    }
  }
  __syncthreads();   // all waves done reading Xcp before OutS (aliased) writes

  v8f acc[4];
#pragma unroll
  for (int nt = 0; nt < 4; ++nt)
#pragma unroll
    for (int i = 0; i < 8; ++i) acc[nt][i] = 0.0f;

  for (int k = 0; k < KK; ++k) {
    v8f z[4];
#pragma unroll
    for (int nt = 0; nt < 4; ++nt)
#pragma unroll
      for (int i = 0; i < 8; ++i) z[nt][i] = 0.0f;

#pragma unroll
    for (int nt = 0; nt < 4; ++nt) {
      // B = Q^T : B[c][o] = Q[o][c]; lane's column o = nt*16 + lrow.
      // B layout: VGPR r holds K = 16*(r>=4) + 8*khalf + 2*(r&3) (+pair)
      const _Float16* qrow = &Qh[(k * 64 + nt * 16 + lrow) * 72];
#pragma unroll
      for (int kc = 0; kc < 2; ++kc) {
        AFrag bf;
#pragma unroll
        for (int r = 0; r < 8; ++r)
          bf.u[r] = *(const unsigned int*)&qrow[kc * 32 + ((r >> 2) << 4) + (khalf << 3) + ((r & 3) << 1)];
        z[nt] = __builtin_amdgcn_wmma_f32_16x16x32_f16(
            false, afr[kc].h, false, bf.h, (short)0, z[nt], false, false);
      }
    }
    // weighted mixture: C/D row mapping M = vgpr_i + 8*khalf
    float wr[8];
#pragma unroll
    for (int i = 0; i < 8; ++i) wr[i] = wmS[(row0 + i + (khalf << 3)) * 8 + k];
#pragma unroll
    for (int nt = 0; nt < 4; ++nt)
#pragma unroll
      for (int i = 0; i < 8; ++i) acc[nt][i] += wr[i] * z[nt][i];
  }

  // ---- stage result [o][p] in LDS (reuses Xcp region) ----
#pragma unroll
  for (int nt = 0; nt < 4; ++nt) {
    int o = nt * 16 + lrow;
    float bias = pbs[o];
#pragma unroll
    for (int i = 0; i < 8; ++i) {
      int p = row0 + i + (khalf << 3);
      OutS[o * 130 + p] = acc[nt][i] + bias;
    }
  }
  __syncthreads();

  // ---- coalesced store to (B, O, H, W) ----
  float* ob = out + (size_t)b * OO * HWHW + plane;
  for (int i = tid; i < OO * TILE; i += THREADS) {
    int o = i >> 7, p = i & 127;
    ob[(size_t)o * HWHW + p] = OutS[o * 130 + p];
  }
}

// ---------------------------------------------------------------------------
extern "C" void kernel_launch(void* const* d_in, const int* in_sizes, int n_in,
                              void* d_out, int out_size, void* d_ws, size_t ws_size,
                              hipStream_t stream) {
  const float* x    = (const float*)d_in[0];
  const float* gw1  = (const float*)d_in[1];
  const float* gb1  = (const float*)d_in[2];
  const float* gw2  = (const float*)d_in[3];
  const float* gb2  = (const float*)d_in[4];
  const float* pw   = (const float*)d_in[5];
  const float* pb   = (const float*)d_in[6];
  const float* s53A = (const float*)d_in[7];
  const float* s53D = (const float*)d_in[8];
  const float* s97A = (const float*)d_in[9];
  const float* s97D = (const float*)d_in[10];

  _Float16* Q = (_Float16*)d_ws;   // 6*64*64 f16 = 48 KB

  build_q_kernel<<<dim3(KK), dim3(64), 0, stream>>>(pw, s53A, s53D, s97A, s97D, Q);

  const int nblocks = (BB * HWHW) / TILE;   // 1024
  wavelet_main_kernel<<<dim3(nblocks), dim3(THREADS), SMEM_BYTES, stream>>>(
      x, gw1, gb1, gw2, gb2, pb, Q, (float*)d_out);
}
